// DynamicMultiBandLinearGenerator_81484119540414
// MI455X (gfx1250) — compile-verified, matrix operand order checked by reading, not run on hardware
//
#include <hip/hip_runtime.h>
#include <hip/hip_bf16.h>
#include <math.h>

// ---------------------------------------------------------------------------
// DynamicMultiBandLinearGenerator for MI455X (gfx1250, wave32, WMMA)
//
// Inputs (all float32), in setup_inputs() order:
//  0:x(16,5,256,256) 1:enc_w1(32,5,3,3) 2:enc_b1 3:enc_w2(64,32,3,3) 4:enc_b2
//  5:enc_w3(64,64,3,3) 6:enc_b3 7:fc_w(805,64) 8:fc_b(805)
//  9:bw0(5,32,1,7,7) 10:bw1(5,32,32,5,5) 11:bw2(5,32,32,3,3)
// 12:bw3(5,32,32,1,1) 13:bw4(5,32,32,1,1) 14:bw5(5,1,32,1,1)
// Output: float32 (16,5,32,32) = 81920 elements.
// ---------------------------------------------------------------------------

typedef _Float16 h8   __attribute__((ext_vector_type(8)));
typedef _Float16 v16h __attribute__((ext_vector_type(16)));
typedef float    v8f  __attribute__((ext_vector_type(8)));

#define NB   16          // batch
#define HW   256         // spatial
#define NSC  805         // total scales per sample (5*161)
#define PERB 161         // scales per band

__device__ __forceinline__ int mirror256(int i) {
    return i < 0 ? -i : (i > 255 ? 510 - i : i);
}

// ------------------------- encoder: conv1 (5->32, 3x3, pad1, relu) ---------
__global__ __launch_bounds__(256) void enc1_kernel(
    const float* __restrict__ x, const float* __restrict__ w,
    const float* __restrict__ bias, _Float16* __restrict__ out)
{
    int idx = blockIdx.x * 256 + threadIdx.x;        // 16*32*256*256 = 1<<25
    int px = idx & 255, py = (idx >> 8) & 255;
    int oc = (idx >> 16) & 31, b = idx >> 21;
    float acc = bias[oc];
    #pragma unroll
    for (int ic = 0; ic < 5; ++ic) {
        const float* xp = x + (((size_t)(b * 5 + ic)) << 16);
        const float* wp = w + (oc * 5 + ic) * 9;
        #pragma unroll
        for (int kh = 0; kh < 3; ++kh) {
            int sy = py + kh - 1;
            if (sy < 0 || sy > 255) continue;
            #pragma unroll
            for (int kw = 0; kw < 3; ++kw) {
                int sx = px + kw - 1;
                if (sx < 0 || sx > 255) continue;
                acc += xp[sy * 256 + sx] * wp[kh * 3 + kw];
            }
        }
    }
    out[idx] = (_Float16)fmaxf(acc, 0.0f);
}

// ------------------ encoder: stride-2 3x3 conv, pad1, relu (templated) -----
template <int CIN, int COUT, int HIN>
__global__ __launch_bounds__(256) void enc_s2_kernel(
    const _Float16* __restrict__ in, const float* __restrict__ w,
    const float* __restrict__ bias, _Float16* __restrict__ out)
{
    constexpr int HOUT = HIN / 2;
    int idx = blockIdx.x * 256 + threadIdx.x;     // NB*COUT*HOUT*HOUT
    int px = idx & (HOUT - 1);
    int py = (idx / HOUT) & (HOUT - 1);
    int oc = (idx / (HOUT * HOUT)) & (COUT - 1);
    int b  = idx / (HOUT * HOUT * COUT);
    float acc = bias[oc];
    for (int ic = 0; ic < CIN; ++ic) {
        const _Float16* ip = in + (size_t)(b * CIN + ic) * HIN * HIN;
        const float* wp = w + (size_t)(oc * CIN + ic) * 9;
        #pragma unroll
        for (int kh = 0; kh < 3; ++kh) {
            int sy = py * 2 + kh - 1;
            if (sy < 0 || sy >= HIN) continue;
            #pragma unroll
            for (int kw = 0; kw < 3; ++kw) {
                int sx = px * 2 + kw - 1;
                if (sx < 0 || sx >= HIN) continue;
                acc += (float)ip[sy * HIN + sx] * wp[kh * 3 + kw];
            }
        }
    }
    out[idx] = (_Float16)fmaxf(acc, 0.0f);
}

// ---------------------- encoder: global mean over 64x64 --------------------
__global__ __launch_bounds__(256) void pool_mean_kernel(
    const _Float16* __restrict__ in, float* __restrict__ feat)
{
    __shared__ float red[256];
    int bc = blockIdx.x;                             // b*64 + c, 1024 blocks
    const _Float16* p = in + (size_t)bc * 4096;
    float s = 0.f;
    for (int i = threadIdx.x; i < 4096; i += 256) s += (float)p[i];
    red[threadIdx.x] = s;
    __syncthreads();
    for (int off = 128; off > 0; off >>= 1) {
        if (threadIdx.x < off) red[threadIdx.x] += red[threadIdx.x + off];
        __syncthreads();
    }
    if (threadIdx.x == 0) feat[bc] = red[0] * (1.0f / 4096.0f);
}

// ---------------------- FC + tanh -> per-sample scales ---------------------
__global__ __launch_bounds__(256) void fc_scales_kernel(
    const float* __restrict__ feat, const float* __restrict__ w,
    const float* __restrict__ bias, float* __restrict__ scales)
{
    int idx = blockIdx.x * 256 + threadIdx.x;        // 16*805
    if (idx >= NB * NSC) return;
    int j = idx % NSC, b = idx / NSC;
    float acc = bias[j];
    const float* f  = feat + b * 64;
    const float* wp = w + (size_t)j * 64;
    #pragma unroll 8
    for (int c = 0; c < 64; ++c) acc += f[c] * wp[c];
    scales[idx] = 1.0f + 0.1f * tanhf(acc);
}

// --------------- repack band weights into per-lane WMMA-B layout -----------
// wrep element index = (((band*36 + tapG)*2 + nt)*32 + lane)*16 + j
// lane n<16 holds column oc=nt*16+n, K = {g*8+j, 16+g*8+(j-8)} with g=lane>>4.
__global__ __launch_bounds__(256) void repack_kernel(
    const float* __restrict__ bw1, const float* __restrict__ bw2,
    const float* __restrict__ bw3, const float* __restrict__ bw4,
    _Float16* __restrict__ wrep)
{
    int idx = blockIdx.x * 256 + threadIdx.x;        // 5*36*1024 = 184320
    if (idx >= 5 * 36 * 1024) return;
    int j    = idx & 15;
    int lane = (idx >> 4) & 31;
    int nt   = (idx >> 9) & 1;
    int q    = idx >> 10;                            // band*36 + tapG
    int band = q / 36;
    int tapG = q % 36;
    int k, t;
    const float* bw;
    if (tapG < 25)      { bw = bw1; k = 5; t = tapG; }
    else if (tapG < 34) { bw = bw2; k = 3; t = tapG - 25; }
    else if (tapG == 34){ bw = bw3; k = 1; t = 0; }
    else                { bw = bw4; k = 1; t = 0; }
    int oc = nt * 16 + (lane & 15);
    int g  = lane >> 4;
    int ic = (j < 8) ? (g * 8 + j) : (16 + g * 8 + (j - 8));
    float v = bw[(size_t)(((band * 32 + oc) * 32 + ic)) * k * k + t];
    wrep[idx] = (_Float16)v;
}

// ---------------- band layer 0: 7x7, 1->32, reflect pad, *scale ------------
__global__ __launch_bounds__(256) void band_layer0_kernel(
    const float* __restrict__ x, const float* __restrict__ w /* bw0+band*32*49 */,
    const float* __restrict__ scales, int band, _Float16* __restrict__ out)
{
    __shared__ float wl[32 * 49];
    for (int i = threadIdx.x; i < 32 * 49; i += 256) wl[i] = w[i];
    __syncthreads();
    int idx = blockIdx.x * 256 + threadIdx.x;        // 16*256*256 = 1<<20
    int px = idx & 255, py = (idx >> 8) & 255, b = idx >> 16;
    const float* xp = x + ((size_t)(b * 5 + band)) * 65536;
    float v[49];
    int t = 0;
    #pragma unroll
    for (int kh = 0; kh < 7; ++kh) {
        int sy = mirror256(py + kh - 3);
        #pragma unroll
        for (int kw = 0; kw < 7; ++kw) {
            int sx = mirror256(px + kw - 3);
            v[t++] = xp[sy * 256 + sx];
        }
    }
    const float* sp = scales + b * NSC + band * PERB;
    h8 r[4];
    for (int oc = 0; oc < 32; ++oc) {
        float a = 0.f;
        #pragma unroll
        for (int tt = 0; tt < 49; ++tt) a += v[tt] * wl[oc * 49 + tt];
        a *= sp[oc];
        r[oc >> 3][oc & 7] = (_Float16)a;
    }
    _Float16* op = out + (size_t)idx * 32;
    *(h8*)(op + 0)  = r[0];
    *(h8*)(op + 8)  = r[1];
    *(h8*)(op + 16) = r[2];
    *(h8*)(op + 24) = r[3];
}

// ------------- WMMA implicit-GEMM conv: 32->32 ch, reflect pad -------------
// NHWC f16 activations. Each wave: 16-pixel row strip x 32 oc (2 WMMA tiles
// per K-step of 32 input channels). KS*KS taps. Scales fused, LDS transpose
// for coalesced f16 output stores.
template <int KS>
__global__ __launch_bounds__(256) void wmma_conv_kernel(
    const _Float16* __restrict__ in, _Float16* __restrict__ out,
    const _Float16* __restrict__ wrep,   // this band+layer's tap blocks
    const float* __restrict__ scales, int scale_off /* band*161+32*li */)
{
    constexpr int P = KS / 2;
    __shared__ float lds[8 * 16 * 32];               // 2KB per wave
    const int lane  = threadIdx.x & 31;
    const int wv    = threadIdx.x >> 5;
    const int strip = blockIdx.x * 8 + wv;           // 65536 strips
    const int sx16  = strip & 15;
    const int y     = (strip >> 4) & 255;
    const int b     = strip >> 12;
    const int n     = lane & 15;                     // pixel (A) / column (C)
    const int g     = lane >> 4;
    const int x     = sx16 * 16 + n;

    v8f c0 = {};
    v8f c1 = {};
    #pragma unroll
    for (int kh = 0; kh < KS; ++kh) {
        const int sy = mirror256(y + kh - P);
        #pragma unroll
        for (int kw = 0; kw < KS; ++kw) {
            const int sx = mirror256(x + kw - P);
            const _Float16* ap =
                in + ((((size_t)b << 8) + sy) * 256 + sx) * 32 + g * 8;
            union { v16h v; struct { h8 lo, hi; } s; } A, B0, B1;
            A.s.lo = *(const h8*)(ap);
            A.s.hi = *(const h8*)(ap + 16);
            const _Float16* bp = wrep + (size_t)(kh * KS + kw) * 1024 + lane * 16;
            B0.s.lo = *(const h8*)(bp);
            B0.s.hi = *(const h8*)(bp + 8);
            B1.s.lo = *(const h8*)(bp + 512);
            B1.s.hi = *(const h8*)(bp + 520);
            c0 = __builtin_amdgcn_wmma_f32_16x16x32_f16(
                     false, A.v, false, B0.v, (short)0, c0, false, false);
            c1 = __builtin_amdgcn_wmma_f32_16x16x32_f16(
                     false, A.v, false, B1.v, (short)0, c1, false, false);
        }
    }

    // fused per-(sample, out-channel) scale: lane's column is fixed.
    const float s0 = scales[b * NSC + scale_off + n];
    const float s1 = scales[b * NSC + scale_off + 16 + n];
    c0 *= s0;
    c1 *= s1;

    // LDS transpose: C layout -> pixel-major rows.
    float* L = lds + wv * 512;
    #pragma unroll
    for (int r = 0; r < 8; ++r) {
        L[(g * 8 + r) * 32 + n]      = c0[r];
        L[(g * 8 + r) * 32 + 16 + n] = c1[r];
    }
    // each lane writes 16 contiguous channels of one pixel.
    const float* Lr = lds + wv * 512 + n * 32 + g * 16;
    float4 f0 = *(const float4*)(Lr + 0);
    float4 f1 = *(const float4*)(Lr + 4);
    float4 f2 = *(const float4*)(Lr + 8);
    float4 f3 = *(const float4*)(Lr + 12);
    h8 lo = { (_Float16)f0.x, (_Float16)f0.y, (_Float16)f0.z, (_Float16)f0.w,
              (_Float16)f1.x, (_Float16)f1.y, (_Float16)f1.z, (_Float16)f1.w };
    h8 hi = { (_Float16)f2.x, (_Float16)f2.y, (_Float16)f2.z, (_Float16)f2.w,
              (_Float16)f3.x, (_Float16)f3.y, (_Float16)f3.z, (_Float16)f3.w };
    _Float16* op =
        out + ((((size_t)b << 8) + y) * 256 + sx16 * 16 + n) * 32 + g * 16;
    *(h8*)(op + 0) = lo;
    *(h8*)(op + 8) = hi;
}

// ------- band tail: 1x1 conv 32->1, *scale, fused 8x8 average pool ---------
__global__ __launch_bounds__(256) void band_tail_kernel(
    const _Float16* __restrict__ act, const float* __restrict__ w5 /* +band*32 */,
    const float* __restrict__ scales, int band, float* __restrict__ outp)
{
    int idx = blockIdx.x * 256 + threadIdx.x;        // 16*32*32 = 16384
    int ox = idx & 31, oy = (idx >> 5) & 31, b = idx >> 10;
    const float s = scales[b * NSC + band * PERB + 160];
    float acc = 0.f;
    for (int yy = 0; yy < 8; ++yy) {
        for (int xx = 0; xx < 8; ++xx) {
            const _Float16* p =
                act + (((size_t)(b * 256 + oy * 8 + yy)) * 256 + ox * 8 + xx) * 32;
            const h8* hp = (const h8*)p;
            float d = 0.f;
            #pragma unroll
            for (int cc = 0; cc < 4; ++cc) {
                h8 hv = hp[cc];
                #pragma unroll
                for (int j = 0; j < 8; ++j) d += (float)hv[j] * w5[cc * 8 + j];
            }
            acc += d * s;
        }
    }
    outp[((size_t)(b * 5 + band) << 10) + (oy << 5) + ox] = acc * (1.0f / 64.0f);
}

// ---------------------------------------------------------------------------
extern "C" void kernel_launch(void* const* d_in, const int* in_sizes, int n_in,
                              void* d_out, int out_size, void* d_ws, size_t ws_size,
                              hipStream_t stream) {
    const float* x      = (const float*)d_in[0];
    const float* enc_w1 = (const float*)d_in[1];
    const float* enc_b1 = (const float*)d_in[2];
    const float* enc_w2 = (const float*)d_in[3];
    const float* enc_b2 = (const float*)d_in[4];
    const float* enc_w3 = (const float*)d_in[5];
    const float* enc_b3 = (const float*)d_in[6];
    const float* fc_w   = (const float*)d_in[7];
    const float* fc_b   = (const float*)d_in[8];
    const float* bw0    = (const float*)d_in[9];
    const float* bw1    = (const float*)d_in[10];
    const float* bw2    = (const float*)d_in[11];
    const float* bw3    = (const float*)d_in[12];
    const float* bw4    = (const float*)d_in[13];
    const float* bw5    = (const float*)d_in[14];
    float* outp = (float*)d_out;

    // workspace layout (needs ~134.6 MB):
    //   [0, 64MB)      ping  (also encoder conv1 output, f16 NCHW)
    //   [64MB,128MB)   pong  (also encoder conv2 @ +0, conv3 @ +32MB, f16 NCHW)
    //   [128MB, ...)   misc: scales(51.5KB) | feat(4KB) | wrep(360KB)
    const size_t ACT_BYTES = (size_t)NB * HW * HW * 32 * 2;   // 67,108,864
    _Float16* ping  = (_Float16*)d_ws;
    _Float16* pong  = (_Float16*)((char*)d_ws + ACT_BYTES);
    char*     misc  = (char*)d_ws + 2 * ACT_BYTES;
    float*    scales = (float*)misc;                          // 16*805 f32
    float*    feat   = (float*)(misc + 65536);                // 16*64 f32
    _Float16* wrep   = (_Float16*)(misc + 131072);            // 184320 f16
    _Float16* enc1o  = ping;
    _Float16* enc2o  = pong;
    _Float16* enc3o  = (_Float16*)((char*)d_ws + ACT_BYTES + 33554432);

    // ---- encoder -> per-sample scales ----
    enc1_kernel<<<131072, 256, 0, stream>>>(x, enc_w1, enc_b1, enc1o);
    enc_s2_kernel<32, 64, 256><<<65536, 256, 0, stream>>>(enc1o, enc_w2, enc_b2, enc2o);
    enc_s2_kernel<64, 64, 128><<<16384, 256, 0, stream>>>(enc2o, enc_w3, enc_b3, enc3o);
    pool_mean_kernel<<<1024, 256, 0, stream>>>(enc3o, feat);
    fc_scales_kernel<<<(NB * NSC + 255) / 256, 256, 0, stream>>>(feat, fc_w, fc_b, scales);

    // ---- repack WMMA weights for layers 1..4, all bands ----
    repack_kernel<<<720, 256, 0, stream>>>(bw1, bw2, bw3, bw4, wrep);

    // ---- per-band filter stacks ----
    for (int band = 0; band < 5; ++band) {
        const _Float16* wb = wrep + (size_t)band * 36 * 1024;
        band_layer0_kernel<<<4096, 256, 0, stream>>>(
            x, bw0 + (size_t)band * 32 * 49, scales, band, ping);
        wmma_conv_kernel<5><<<8192, 256, 0, stream>>>(
            ping, pong, wb +  0 * 1024, scales, band * PERB + 32 * 1);
        wmma_conv_kernel<3><<<8192, 256, 0, stream>>>(
            pong, ping, wb + 25 * 1024, scales, band * PERB + 32 * 2);
        wmma_conv_kernel<1><<<8192, 256, 0, stream>>>(
            ping, pong, wb + 34 * 1024, scales, band * PERB + 32 * 3);
        wmma_conv_kernel<1><<<8192, 256, 0, stream>>>(
            pong, ping, wb + 35 * 1024, scales, band * PERB + 32 * 4);
        band_tail_kernel<<<64, 256, 0, stream>>>(
            ping, bw5 + (size_t)band * 32, scales, band, outp);
    }
}
